// Block_26611617366637
// MI455X (gfx1250) — matrix operand order repaired and checked by reading, not compiled, not run on hardware
//
#include <hip/hip_runtime.h>
#include <hip/hip_bf16.h>

typedef __bf16 bf16;
typedef __attribute__((ext_vector_type(16))) __bf16 bf16x16;
typedef __attribute__((ext_vector_type(8)))  float   f32x8;

#define TILE_BM 128
#define TILE_BN 128
#define TILE_BK 32
#define LDS_STRIDE 40   // bf16 elements; 80B rows keep 16B alignment, skew banks

union Frag16 { uint4 q[2]; bf16x16 v; };

// ---------------------------------------------------------------------------
// CDNA5 async global->LDS copy (guarded; falls back to sync load+ds_store)
// ---------------------------------------------------------------------------
#if defined(__has_builtin)
#if __has_builtin(__builtin_amdgcn_global_load_async_to_lds_b128) && \
    __has_builtin(__builtin_amdgcn_s_wait_asynccnt)
#define USE_ASYNC_LDS 1
#endif
#endif

typedef int v4i __attribute__((vector_size(16)));
typedef __attribute__((address_space(1))) v4i GlobalV4;
typedef __attribute__((address_space(3))) v4i LdsV4;

__device__ __forceinline__ void copy16B(void* lds, const void* g) {
#ifdef USE_ASYNC_LDS
  __builtin_amdgcn_global_load_async_to_lds_b128(
      (GlobalV4*)g, (LdsV4*)lds, 0, 0);
#else
  *reinterpret_cast<uint4*>(lds) = *reinterpret_cast<const uint4*>(g);
#endif
}

__device__ __forceinline__ void wait_async_le4() {
#ifdef USE_ASYNC_LDS
  __builtin_amdgcn_s_wait_asynccnt(4);   // 4 newer loads may still be in flight
#endif
}
__device__ __forceinline__ void wait_async_0() {
#ifdef USE_ASYNC_LDS
  __builtin_amdgcn_s_wait_asynccnt(0);
#endif
}

// ---------------------------------------------------------------------------
// Weight prep: fp32 [R,C] -> bf16 transposed [C,R]
// ---------------------------------------------------------------------------
__global__ __launch_bounds__(256) void k_cast_transpose_f32(
    const float* __restrict__ in, bf16* __restrict__ out, int R, int C) {
  long i = (long)blockIdx.x * 256 + threadIdx.x;
  if (i >= (long)R * C) return;
  int orow = (int)(i / R);   // output row = input col
  int ocol = (int)(i % R);   // output col = input row
  out[i] = (bf16)in[(long)ocol * C + orow];
}

// bf16 [R,C] -> bf16 [C,R], batched via blockIdx.y
__global__ __launch_bounds__(256) void k_transpose_bf16(
    const bf16* __restrict__ in, bf16* __restrict__ out, int R, int C) {
  long base = (long)blockIdx.y * (long)R * C;
  long i = (long)blockIdx.x * 256 + threadIdx.x;
  if (i >= (long)R * C) return;
  int orow = (int)(i / R);
  int ocol = (int)(i % R);
  out[base + i] = in[base + (long)ocol * C + orow];
}

// ---------------------------------------------------------------------------
// LayerNorm: one row per block, fp32 in -> bf16 out
// ---------------------------------------------------------------------------
__global__ __launch_bounds__(256) void k_layernorm(
    const float* __restrict__ x, const float* __restrict__ gamma,
    const float* __restrict__ beta, bf16* __restrict__ out, int C) {
  long row = blockIdx.x;
  const float* xr = x + row * C;
  int tid = threadIdx.x;
  float s = 0.f, s2 = 0.f;
  for (int c = tid; c < C; c += 256) { float v = xr[c]; s += v; s2 += v * v; }
  __shared__ float sh[256], sh2[256];
  sh[tid] = s; sh2[tid] = s2; __syncthreads();
  for (int o = 128; o > 0; o >>= 1) {
    if (tid < o) { sh[tid] += sh[tid + o]; sh2[tid] += sh2[tid + o]; }
    __syncthreads();
  }
  float mean = sh[0] / C;
  float var  = sh2[0] / C - mean * mean;
  float inv  = rsqrtf(var + 1e-5f);
  for (int c = tid; c < C; c += 256)
    out[row * C + c] = (bf16)((xr[c] - mean) * inv * gamma[c] + beta[c]);
}

// ---------------------------------------------------------------------------
// Causal softmax over one row (b, t): fp32 scores -> bf16 probs, P[j>t]=0
// ---------------------------------------------------------------------------
__global__ __launch_bounds__(256) void k_softmax_causal(
    const float* __restrict__ S, bf16* __restrict__ P, int T) {
  int t = blockIdx.x;
  long base = ((long)blockIdx.y * T + t) * T;
  const float* Sr = S + base;
  bf16* Pr = P + base;
  int tid = threadIdx.x;
  __shared__ float sh[256];
  float m = -3.4e38f;
  for (int j = tid; j <= t; j += 256) m = fmaxf(m, Sr[j]);
  sh[tid] = m; __syncthreads();
  for (int o = 128; o > 0; o >>= 1) {
    if (tid < o) sh[tid] = fmaxf(sh[tid], sh[tid + o]);
    __syncthreads();
  }
  m = sh[0]; __syncthreads();
  float s = 0.f;
  for (int j = tid; j <= t; j += 256) s += __expf(Sr[j] - m);
  sh[tid] = s; __syncthreads();
  for (int o = 128; o > 0; o >>= 1) {
    if (tid < o) sh[tid] += sh[tid + o];
    __syncthreads();
  }
  float inv = 1.f / sh[0];
  for (int j = tid; j < T; j += 256)
    Pr[j] = (j <= t) ? (bf16)(__expf(Sr[j] - m) * inv) : (bf16)0.f;
}

// ---------------------------------------------------------------------------
// Tiled bf16 WMMA GEMM:  C[b] = alpha * A[b] @ Bt[b]^T  (+bias)(+resid)(relu)
//   A  : [M,K] bf16 row-major, leading dim lda, batch stride sA
//   Bt : [N,K] bf16 row-major (i.e. B transposed), ldb, batch stride sB
//   C  : [M,N] fp32 or bf16, ldc, batch stride sC (resid shares C layout)
// Block = 256 threads = 8 waves, 2x4 wave grid; wave owns 64x32 (4x2 WMMA
// tiles). LDS double-buffered; staging uses async global->LDS when available.
// ---------------------------------------------------------------------------
template <bool BIAS, bool RELU, bool RESID, bool OUT_BF16, bool CAUSAL>
__global__ __launch_bounds__(256) void k_gemm_bf16(
    const bf16* __restrict__ A, int lda, long sA,
    const bf16* __restrict__ Bt, int ldb, long sB,
    void* __restrict__ Cout, int ldc, long sC,
    const float* __restrict__ bias, const float* __restrict__ resid,
    int K, float alpha) {
  int m0 = blockIdx.y * TILE_BM;
  int n0 = blockIdx.x * TILE_BN;
  if (CAUSAL && n0 > m0 + TILE_BM - 1) return;  // tile entirely above diagonal

  long zA = (long)blockIdx.z * sA;
  long zB = (long)blockIdx.z * sB;
  long zC = (long)blockIdx.z * sC;

  __shared__ __align__(16) bf16 As[2][TILE_BM * LDS_STRIDE];
  __shared__ __align__(16) bf16 Bs[2][TILE_BN * LDS_STRIDE];

  int tid  = threadIdx.x;
  int lane = tid & 31;
  int wave = tid >> 5;
  int wm = wave & 1;          // 2 wave rows of 64
  int wn = wave >> 1;         // 4 wave cols of 32
  int m0w = wm * 64;
  int n0w = wn * 32;
  int g = lane >> 4;          // lane group (ISA 16-bit A/B K-split)
  int r = lane & 15;

  // Staging map: thread -> (row, 16-element segment); 32B contiguous per thread
  int rowLd = tid >> 1;
  int segLd = (tid & 1) * 16;
  const bf16* gA = A + zA + (long)(m0 + rowLd) * lda + segLd;
  const bf16* gB = Bt + zB + (long)(n0 + rowLd) * ldb + segLd;
  const int ldsOff = rowLd * LDS_STRIDE + segLd;

  f32x8 acc[4][2] = {};

  const int nIter = K / TILE_BK;

  // Prologue: stage tile 0 into buffer 0
  copy16B(&As[0][ldsOff],     gA);
  copy16B(&As[0][ldsOff + 8], gA + 8);
  copy16B(&Bs[0][ldsOff],     gB);
  copy16B(&Bs[0][ldsOff + 8], gB + 8);

  for (int it = 0; it < nIter; ++it) {
    int cur = it & 1;
    if (it + 1 < nIter) {
      const bf16* a = gA + (long)(it + 1) * TILE_BK;
      const bf16* b = gB + (long)(it + 1) * TILE_BK;
      copy16B(&As[cur ^ 1][ldsOff],     a);
      copy16B(&As[cur ^ 1][ldsOff + 8], a + 8);
      copy16B(&Bs[cur ^ 1][ldsOff],     b);
      copy16B(&Bs[cur ^ 1][ldsOff + 8], b + 8);
      wait_async_le4();   // current tile's 4 loads are complete
    } else {
      wait_async_0();
    }
    __syncthreads();      // all waves' staging for `cur` visible

    bf16x16 af[4];
    bf16x16 bfv[2];
    #pragma unroll
    for (int mi = 0; mi < 4; ++mi) {
      // A 16-bit layout: lanes 0-15 hold K=0..7 & 16..23; lanes 16-31 K=8..15 & 24..31
      const bf16* p = &As[cur][(m0w + mi * 16 + r) * LDS_STRIDE + g * 8];
      Frag16 f;
      f.q[0] = *reinterpret_cast<const uint4*>(p);
      f.q[1] = *reinterpret_cast<const uint4*>(p + 16);
      af[mi] = f.v;
    }
    #pragma unroll
    for (int ni = 0; ni < 2; ++ni) {
      // B 16-bit layout: lanes 0-15 hold K=0..7 & 16..23; lanes 16-31 K=8..15 & 24..31
      const bf16* p = &Bs[cur][(n0w + ni * 16 + r) * LDS_STRIDE + g * 8];
      Frag16 f;
      f.q[0] = *reinterpret_cast<const uint4*>(p);
      f.q[1] = *reinterpret_cast<const uint4*>(p + 16);
      bfv[ni] = f.v;
    }
    #pragma unroll
    for (int mi = 0; mi < 4; ++mi)
      #pragma unroll
      for (int ni = 0; ni < 2; ++ni)
        acc[mi][ni] = __builtin_amdgcn_wmma_f32_16x16x32_bf16(
            false, af[mi], false, bfv[ni], (short)0, acc[mi][ni], false, false);

    __syncthreads();      // reads of `cur` done before it is restaged
  }

  // Epilogue: C/D layout — VGPR e: lanes 0-15 -> M=e, lanes 16-31 -> M=8+e
  int colLane = lane & 15;
  int rowOff  = (lane >> 4) * 8;
  #pragma unroll
  for (int mi = 0; mi < 4; ++mi) {
    #pragma unroll
    for (int ni = 0; ni < 2; ++ni) {
      int mt = m0 + m0w + mi * 16;
      int nt = n0 + n0w + ni * 16;
      #pragma unroll
      for (int e = 0; e < 8; ++e) {
        int row = mt + rowOff + e;
        int col = nt + colLane;
        long idx = zC + (long)row * ldc + col;
        float v = acc[mi][ni][e] * alpha;
        if (BIAS)  v += bias[col];
        if (RESID) v += resid[idx];
        if (RELU)  v = fmaxf(v, 0.f);
        if (OUT_BF16) reinterpret_cast<bf16*>(Cout)[idx] = (bf16)v;
        else          reinterpret_cast<float*>(Cout)[idx] = v;
      }
    }
  }
}

// ---------------------------------------------------------------------------
extern "C" void kernel_launch(void* const* d_in, const int* in_sizes, int n_in,
                              void* d_out, int out_size, void* d_ws, size_t ws_size,
                              hipStream_t stream) {
  (void)in_sizes; (void)n_in; (void)out_size; (void)ws_size;
  const int B = 4, T = 2048, C = 1024, H = 1024, F = 4096;
  const long BT = (long)B * T;          // 8192 rows
  const long TH = (long)T * H;          // per-batch q/k/v stride
  const long TT = (long)T * T;          // per-batch score stride

  const float* x   = (const float*)d_in[0];
  const float* Wk  = (const float*)d_in[1];
  const float* Wq  = (const float*)d_in[2];
  const float* Wv  = (const float*)d_in[3];
  const float* W1  = (const float*)d_in[4];
  const float* b1  = (const float*)d_in[5];
  const float* W2  = (const float*)d_in[6];
  const float* b2  = (const float*)d_in[7];
  const float* g1  = (const float*)d_in[8];
  const float* be1 = (const float*)d_in[9];
  const float* g2  = (const float*)d_in[10];
  const float* be2 = (const float*)d_in[11];

  // Workspace carve-up (256B aligned)
  char* ws = (char*)d_ws;
  size_t off = 0;
  auto alloc = [&](size_t bytes) {
    void* p = ws + off;
    off += (bytes + 255) & ~(size_t)255;
    return p;
  };
  bf16* WqT = (bf16*)alloc((size_t)C * H * 2);
  bf16* WkT = (bf16*)alloc((size_t)C * H * 2);
  bf16* WvT = (bf16*)alloc((size_t)C * H * 2);
  bf16* W1T = (bf16*)alloc((size_t)C * F * 2);
  bf16* W2T = (bf16*)alloc((size_t)F * C * 2);
  bf16* h   = (bf16*)alloc((size_t)BT * C * 2);
  bf16* q   = (bf16*)alloc((size_t)BT * H * 2);
  bf16* k   = (bf16*)alloc((size_t)BT * H * 2);
  bf16* v   = (bf16*)alloc((size_t)BT * H * 2);
  bf16* vT  = (bf16*)alloc((size_t)B * H * T * 2);
  float* S  = (float*)alloc((size_t)B * T * T * 4);
  bf16* P   = (bf16*)alloc((size_t)B * T * T * 2);
  float* x1 = (float*)alloc((size_t)BT * H * 4);
  bf16* h2  = (bf16*)alloc((size_t)BT * C * 2);
  bf16* act = (bf16*)alloc((size_t)BT * F * 2);

  // 1) Weights -> bf16, transposed
  k_cast_transpose_f32<<<(C * H) / 256, 256, 0, stream>>>(Wq, WqT, C, H);
  k_cast_transpose_f32<<<(C * H) / 256, 256, 0, stream>>>(Wk, WkT, C, H);
  k_cast_transpose_f32<<<(C * H) / 256, 256, 0, stream>>>(Wv, WvT, C, H);
  k_cast_transpose_f32<<<(C * F) / 256, 256, 0, stream>>>(W1, W1T, C, F);
  k_cast_transpose_f32<<<(F * C) / 256, 256, 0, stream>>>(W2, W2T, F, C);

  // 2) h = LN(x; g1, be1)
  k_layernorm<<<(int)BT, 256, 0, stream>>>(x, g1, be1, h, C);

  // 3) q/k/v = h @ W{q,k,v}
  dim3 gQKV(H / TILE_BN, (int)(BT / TILE_BM), 1);
  k_gemm_bf16<false, false, false, true, false><<<gQKV, 256, 0, stream>>>(
      h, C, 0, WqT, C, 0, q, H, 0, nullptr, nullptr, C, 1.0f);
  k_gemm_bf16<false, false, false, true, false><<<gQKV, 256, 0, stream>>>(
      h, C, 0, WkT, C, 0, k, H, 0, nullptr, nullptr, C, 1.0f);
  k_gemm_bf16<false, false, false, true, false><<<gQKV, 256, 0, stream>>>(
      h, C, 0, WvT, C, 0, v, H, 0, nullptr, nullptr, C, 1.0f);

  // 4) vT[b] = v[b]^T  (so P@V GEMM sees Bt row-major)
  dim3 gTr((T * H) / 256, B, 1);
  k_transpose_bf16<<<gTr, 256, 0, stream>>>(v, vT, T, H);

  // 5) S[b] = (q[b] @ k[b]^T) / sqrt(C)   (Bt = k itself); causal tiles skipped
  dim3 gS(T / TILE_BN, T / TILE_BM, B);
  k_gemm_bf16<false, false, false, false, true><<<gS, 256, 0, stream>>>(
      q, H, TH, k, H, TH, S, T, TT, nullptr, nullptr, H, 0.03125f);

  // 6) P = causal_softmax(S)
  dim3 gSm(T, B, 1);
  k_softmax_causal<<<gSm, 256, 0, stream>>>(S, P, T);

  // 7) x1 = x + P @ V
  dim3 gPV(H / TILE_BN, T / TILE_BM, B);
  k_gemm_bf16<false, false, true, false, false><<<gPV, 256, 0, stream>>>(
      P, T, TT, vT, T, (long)H * T, x1, H, TH, nullptr, x, T, 1.0f);

  // 8) h2 = LN(x1; g2, be2)
  k_layernorm<<<(int)BT, 256, 0, stream>>>(x1, g2, be2, h2, C);

  // 9) act = relu(h2 @ W1 + b1)
  dim3 gF1(F / TILE_BN, (int)(BT / TILE_BM), 1);
  k_gemm_bf16<true, true, false, true, false><<<gF1, 256, 0, stream>>>(
      h2, C, 0, W1T, C, 0, act, F, 0, b1, nullptr, C, 1.0f);

  // 10) out = x1 + act @ W2 + b2
  dim3 gF2(C / TILE_BN, (int)(BT / TILE_BM), 1);
  k_gemm_bf16<true, false, true, false, false><<<gF2, 256, 0, stream>>>(
      act, F, 0, W2T, F, 0, (float*)d_out, C, 0, b2, x1, F, 1.0f);
}
